// CrossAttentionFusion_34385508172536
// MI455X (gfx1250) — matrix operand order repaired and checked by reading, not compiled, not run on hardware
//
#include <hip/hip_runtime.h>
#include <hip/hip_bf16.h>

// ---------------------------------------------------------------------------
// CrossAttentionFusion for MI455X (gfx1250, wave32, WMMA).
// One wave = one 16-row tile, full pipeline fused, all intermediates in LDS.
// All GEMMs via v_wmma_f32_16x16x32_f16 (f16 in, f32 accumulate).
// Weights pre-converted f32->f16 into d_ws once per launch (L2-resident).
// ---------------------------------------------------------------------------

typedef __attribute__((ext_vector_type(16))) _Float16 v16h;
typedef __attribute__((ext_vector_type(8)))  _Float16 h8;
typedef __attribute__((ext_vector_type(8)))  float    v8f;

#define SLD 1048   // arena row stride in halves: 16B-aligned, 12-bank row skew
                   // -> the 16 row-parallel b128 A-frag reads are conflict-free
#define C_A   0    // a / r_a / h1 / h(gelu)
#define C_T   256  // t / r_t / h2
#define C_XA  512  // v_a2t / a_out (x[:, :256])
#define C_XT  768  // v_t2a / t_out (x[:, 256:])

// f16 weight arena offsets in d_ws (element counts)
#define OFF_WA      0         // [256][256]
#define OFF_WT      65536     // [256][768]
#define OFF_WV_A2T  262144    // [256][256]  rows 512..767 of a2t_in_w
#define OFF_WO_A2T  327680    // [256][256]
#define OFF_WV_T2A  393216    // [256][256]
#define OFF_WO_T2A  458752    // [256][256]
#define OFF_W1      524288    // [256][512]
#define OFF_W2      655360    // [128][256]
#define OFF_W3      688128    // [16][128]  (rows 7..15 zero-padded)

__device__ __forceinline__ float gelu_exact(float x) {
  return 0.5f * x * (1.0f + erff(x * 0.70710678118654752f));
}

// A-fragment (16x32 f16) from LDS arena; lane m=l&15 holds row m.
// halves 0..7  -> k = kb + 8*hi + i ; halves 8..15 -> k = kb + 16 + 8*hi + i
__device__ __forceinline__ v16h lds_load_a(const _Float16* base, int kb, int lane) {
  const int m = lane & 15, hi = lane >> 4;
  const _Float16* p = base + m * SLD + kb + hi * 8;
  h8 lo = *(const h8*)(p);
  h8 hh = *(const h8*)(p + 16);
  return __builtin_shufflevector(lo, hh, 0,1,2,3,4,5,6,7,8,9,10,11,12,13,14,15);
}

// B-fragment (32x16 f16): B = W^T, so lane n reads 16 f16 of W row n (two runs of 8).
// halves 0..7 -> k = kb + 8*hi + i ; halves 8..15 -> k = kb + 16 + 8*hi + i
__device__ __forceinline__ v16h gl_load_b(const _Float16* __restrict__ W, int K,
                                          int n0, int kb, int lane) {
  const int n = n0 + (lane & 15), hi = lane >> 4;
  const _Float16* p = W + (size_t)n * K + kb + hi * 8;
  h8 lo = *(const h8*)(p);
  h8 hh = *(const h8*)(p + 16);
  return __builtin_shufflevector(lo, hh, 0,1,2,3,4,5,6,7,8,9,10,11,12,13,14,15);
}

// A-fragment from global f32 input (ld == K), converted to f16 in registers.
__device__ __forceinline__ v16h gl_load_a32(const float* __restrict__ A, int ld,
                                            int row0, int kb, int lane) {
  const int m = lane & 15, hi = lane >> 4;
  const float* p = A + (size_t)(row0 + m) * ld + kb + hi * 8;
  v8f x0 = *(const v8f*)(p);
  v8f x1 = *(const v8f*)(p + 16);
  v16h r;
#pragma unroll
  for (int i = 0; i < 8; ++i) { r[i] = (_Float16)x0[i]; r[8 + i] = (_Float16)x1[i]; }
  return r;
}

// D/C layout: lane n + 16*(m>=8); vgpr r -> m = r + 8*hi.
// EPI: 0 = bias+store f16, 1 = bias+residual(in-place)+store, 2 = bias+gelu+store
template<int K, int N, int EPI>
__device__ __forceinline__ void gemm_lds(_Float16* arena, int aCol,
                                         const _Float16* __restrict__ W,
                                         const float* __restrict__ bias,
                                         int oCol, int lane) {
  const int nLoc = lane & 15, hi = lane >> 4;
#pragma unroll 1
  for (int n0 = 0; n0 < N; n0 += 16) {
    v8f acc = {};
#pragma unroll
    for (int kb = 0; kb < K; kb += 32) {
      v16h a = lds_load_a(arena + aCol, kb, lane);
      v16h b = gl_load_b(W, K, n0, kb, lane);
      acc = __builtin_amdgcn_wmma_f32_16x16x32_f16(false, a, false, b,
                                                   (short)0, acc, false, false);
    }
    const int n = n0 + nLoc;
    const float bn = bias[n];
#pragma unroll
    for (int r = 0; r < 8; ++r) {
      _Float16* o = arena + (r + 8 * hi) * SLD + oCol + n;
      float v = acc[r] + bn;
      if (EPI == 1) v += (float)(*o);
      if (EPI == 2) v = gelu_exact(v);
      *o = (_Float16)v;
    }
  }
}

// Same GEMM but with the A operand streamed from a global f32 matrix.
template<int K, int N>
__device__ __forceinline__ void gemm_gl(const float* __restrict__ A, int row0,
                                        const _Float16* __restrict__ W,
                                        const float* __restrict__ bias,
                                        _Float16* arena, int oCol, int lane) {
  const int nLoc = lane & 15, hi = lane >> 4;
#pragma unroll 1
  for (int n0 = 0; n0 < N; n0 += 16) {
    v8f acc = {};
#pragma unroll
    for (int kb = 0; kb < K; kb += 32) {
      v16h a = gl_load_a32(A, K, row0, kb, lane);
      v16h b = gl_load_b(W, K, n0, kb, lane);
      acc = __builtin_amdgcn_wmma_f32_16x16x32_f16(false, a, false, b,
                                                   (short)0, acc, false, false);
    }
    const int n = n0 + nLoc;
    const float bn = bias[n];
#pragma unroll
    for (int r = 0; r < 8; ++r)
      arena[(r + 8 * hi) * SLD + oCol + n] = (_Float16)(acc[r] + bn);
  }
}

// Row-wise LayerNorm over a 16x256 LDS panel, wave-cooperative.
template<bool GELU>
__device__ __forceinline__ void ln_rows(_Float16* arena, int srcc, int dstc,
                                        const float* __restrict__ g,
                                        const float* __restrict__ b, int lane) {
#pragma unroll 1
  for (int m = 0; m < 16; ++m) {
    _Float16* row = arena + m * SLD;
    float s1 = 0.f, s2 = 0.f;
#pragma unroll
    for (int c = lane; c < 256; c += 32) {
      float v = (float)row[srcc + c];
      s1 += v; s2 += v * v;
    }
#pragma unroll
    for (int off = 16; off > 0; off >>= 1) {
      s1 += __shfl_xor(s1, off, 32);
      s2 += __shfl_xor(s2, off, 32);
    }
    const float mean = s1 * (1.f / 256.f);
    const float var  = s2 * (1.f / 256.f) - mean * mean;
    const float inv  = rsqrtf(var + 1e-5f);
#pragma unroll
    for (int c = lane; c < 256; c += 32) {
      float v = (float)row[srcc + c];
      float y = (v - mean) * inv * g[c] + b[c];
      if (GELU) y = gelu_exact(y);
      row[dstc + c] = (_Float16)y;
    }
  }
}

// Final: out[16,7] = h2[16,128] @ W3p^T + b3 (W3p zero-padded to 16 rows)
__device__ __forceinline__ void final7(_Float16* arena,
                                       const _Float16* __restrict__ W3p,
                                       const float* __restrict__ b3,
                                       float* __restrict__ out, int row0, int lane) {
  v8f acc = {};
#pragma unroll
  for (int kb = 0; kb < 128; kb += 32) {
    v16h a = lds_load_a(arena + C_T, kb, lane);
    v16h b = gl_load_b(W3p, 128, 0, kb, lane);
    acc = __builtin_amdgcn_wmma_f32_16x16x32_f16(false, a, false, b,
                                                 (short)0, acc, false, false);
  }
  const int n = lane & 15, hi = lane >> 4;
  if (n < 7) {
    const float bn = b3[n];
#pragma unroll
    for (int r = 0; r < 8; ++r)
      out[(size_t)(row0 + r + 8 * hi) * 7 + n] = acc[r] + bn;
  }
}

__global__ __launch_bounds__(32) void caf_fused_kernel(
    const float* __restrict__ audio, const float* __restrict__ text,
    const _Float16* __restrict__ wf,
    const float* __restrict__ ba, const float* __restrict__ bt,
    const float* __restrict__ a2t_in_b, const float* __restrict__ a2t_out_b,
    const float* __restrict__ t2a_in_b, const float* __restrict__ t2a_out_b,
    const float* __restrict__ ln_a_g, const float* __restrict__ ln_a_b,
    const float* __restrict__ ln_t_g, const float* __restrict__ ln_t_b,
    const float* __restrict__ b1, const float* __restrict__ ln1_g,
    const float* __restrict__ ln1_b, const float* __restrict__ b2,
    const float* __restrict__ b3, float* __restrict__ out) {
  __shared__ _Float16 arena[16 * SLD];       // 33,536 B
  const int lane = threadIdx.x;
  const int row0 = blockIdx.x * 16;

  // S1: a = audio @ Wa^T + ba          -> C_A
  gemm_gl<256, 256>(audio, row0, wf + OFF_WA, ba, arena, C_A, lane);
  // S2: t = text @ Wt^T + bt           -> C_T
  gemm_gl<768, 256>(text, row0, wf + OFF_WT, bt, arena, C_T, lane);
  __syncthreads();
  // S3: v_a2t = t @ Wv_a2t^T + bv      -> C_XA
  gemm_lds<256, 256, 0>(arena, C_T, wf + OFF_WV_A2T, a2t_in_b + 512, C_XA, lane);
  // S4: v_t2a = a @ Wv_t2a^T + bv      -> C_XT
  gemm_lds<256, 256, 0>(arena, C_A, wf + OFF_WV_T2A, t2a_in_b + 512, C_XT, lane);
  __syncthreads();
  // S5: r_a = a + v_a2t @ Wo_a2t^T + bo  (in-place residual on C_A)
  gemm_lds<256, 256, 1>(arena, C_XA, wf + OFF_WO_A2T, a2t_out_b, C_A, lane);
  // S6: r_t = t + v_t2a @ Wo_t2a^T + bo  (in-place residual on C_T)
  gemm_lds<256, 256, 1>(arena, C_XT, wf + OFF_WO_T2A, t2a_out_b, C_T, lane);
  __syncthreads();
  // S7/S8: LN -> x = [a_out | t_out] in C_XA..C_XT (contiguous 512 cols)
  ln_rows<false>(arena, C_A, C_XA, ln_a_g, ln_a_b, lane);
  ln_rows<false>(arena, C_T, C_XT, ln_t_g, ln_t_b, lane);
  __syncthreads();
  // S9: h1 = x @ W1^T + b1             -> C_A (raw)
  gemm_lds<512, 256, 0>(arena, C_XA, wf + OFF_W1, b1, C_A, lane);
  __syncthreads();
  // S9b: h = gelu(LN(h1))              -> C_A (in place)
  ln_rows<true>(arena, C_A, C_A, ln1_g, ln1_b, lane);
  __syncthreads();
  // S10: h2 = gelu(h @ W2^T + b2)      -> C_T[0:128]
  gemm_lds<256, 128, 2>(arena, C_A, wf + OFF_W2, b2, C_T, lane);
  __syncthreads();
  // S11: out = h2 @ W3^T + b3          -> global f32 [B,7]
  final7(arena, wf + OFF_W3, b3, out, row0, lane);
}

// ----- weight precondition: f32 -> f16 into d_ws ---------------------------
__global__ void cvt_f16_kernel(const float* __restrict__ s,
                               _Float16* __restrict__ d, int n) {
  int i = blockIdx.x * blockDim.x + threadIdx.x;
  if (i < n) d[i] = (_Float16)s[i];
}
__global__ void cvt_w3_kernel(const float* __restrict__ s,
                              _Float16* __restrict__ d) {
  int i = blockIdx.x * blockDim.x + threadIdx.x;
  if (i < 16 * 128) d[i] = (i < 7 * 128) ? (_Float16)s[i] : (_Float16)0.f;
}

extern "C" void kernel_launch(void* const* d_in, const int* in_sizes, int n_in,
                              void* d_out, int out_size, void* d_ws, size_t ws_size,
                              hipStream_t stream) {
  (void)in_sizes; (void)n_in; (void)out_size; (void)ws_size;
  const float* audio     = (const float*)d_in[0];
  const float* text      = (const float*)d_in[1];
  const float* Wa        = (const float*)d_in[2];
  const float* ba        = (const float*)d_in[3];
  const float* Wt        = (const float*)d_in[4];
  const float* bt        = (const float*)d_in[5];
  const float* a2t_in_w  = (const float*)d_in[6];
  const float* a2t_in_b  = (const float*)d_in[7];
  const float* a2t_out_w = (const float*)d_in[8];
  const float* a2t_out_b = (const float*)d_in[9];
  const float* t2a_in_w  = (const float*)d_in[10];
  const float* t2a_in_b  = (const float*)d_in[11];
  const float* t2a_out_w = (const float*)d_in[12];
  const float* t2a_out_b = (const float*)d_in[13];
  const float* ln_a_g    = (const float*)d_in[14];
  const float* ln_a_b    = (const float*)d_in[15];
  const float* ln_t_g    = (const float*)d_in[16];
  const float* ln_t_b    = (const float*)d_in[17];
  const float* W1        = (const float*)d_in[18];
  const float* b1        = (const float*)d_in[19];
  const float* ln1_g     = (const float*)d_in[20];
  const float* ln1_b     = (const float*)d_in[21];
  const float* W2        = (const float*)d_in[22];
  const float* b2        = (const float*)d_in[23];
  const float* W3        = (const float*)d_in[24];
  const float* b3        = (const float*)d_in[25];
  float* out    = (float*)d_out;
  _Float16* wf  = (_Float16*)d_ws;

  auto cvt = [&](const float* s, _Float16* d, int n) {
    cvt_f16_kernel<<<(n + 255) / 256, 256, 0, stream>>>(s, d, n);
  };
  cvt(Wa, wf + OFF_WA, 256 * 256);
  cvt(Wt, wf + OFF_WT, 256 * 768);
  cvt(a2t_in_w + 2 * 256 * 256, wf + OFF_WV_A2T, 256 * 256);  // Wv slice
  cvt(a2t_out_w, wf + OFF_WO_A2T, 256 * 256);
  cvt(t2a_in_w + 2 * 256 * 256, wf + OFF_WV_T2A, 256 * 256);  // Wv slice
  cvt(t2a_out_w, wf + OFF_WO_T2A, 256 * 256);
  cvt(W1, wf + OFF_W1, 256 * 512);
  cvt(W2, wf + OFF_W2, 128 * 256);
  cvt_w3_kernel<<<8, 256, 0, stream>>>(W3, wf + OFF_W3);

  caf_fused_kernel<<<65536 / 16, 32, 0, stream>>>(
      audio, text, wf, ba, bt, a2t_in_b, a2t_out_b, t2a_in_b, t2a_out_b,
      ln_a_g, ln_a_b, ln_t_g, ln_t_b, b1, ln1_g, ln1_b, b2, b3, out);
}
